// LIGHTEN_frame_level_15839839387971
// MI455X (gfx1250) — compile-verified, matrix operand order checked by reading, not run on hardware
//
#include <hip/hip_runtime.h>
#include <hip/hip_bf16.h>

typedef __bf16 bf16_t;
typedef bf16_t v16bf __attribute__((ext_vector_type(16)));
typedef float  v8f   __attribute__((ext_vector_type(8)));
typedef unsigned short u16;
typedef unsigned long long u64;

#define B_    256
#define T_    20
#define NOBJ_ 5
#define D_    2048
#define P_    1000
#define PP_   1008
#define INP_  2048
#define H_    512
#define HH_   256

#define KC_     64     // K-chunk (bf16 elements) staged per block iteration
#define APITCH_ 88     // LDS row pitch in bf16 elems = 176B: 16B-aligned, 44-bank stride
                       // -> all 16 rows land on distinct banks (44*r mod 64 distinct)

// ---------- bf16 helpers ----------
__device__ __forceinline__ u16 f2bf(float f) {
    unsigned int x = __float_as_uint(f);
    x += 0x7FFFu + ((x >> 16) & 1u);   // round-to-nearest-even
    return (u16)(x >> 16);
}
__device__ __forceinline__ float bf2f(u16 u) {
    return __uint_as_float(((unsigned int)u) << 16);
}

union Frag { v16bf v; uint4 q[2]; };

// ---------- generic WMMA GEMM: C = act(A[M,K] * W[N,K]^T + bias) ----------
// A bf16 (row r at A + r*strideA), W bf16 row-major [Npad][Kp], Kp % 64 == 0.
// Block = 4 waves sharing one M tile; A tile staged to LDS with async copies
// (ASYNCcnt path), W streamed from global with prefetch. M % 16 == 0 for all calls.
__global__ void wmma_gemm_bf16(const u16* __restrict__ A, u64 strideA,
                               const u16* __restrict__ W, int Kp,
                               const float* __restrict__ bias,
                               float* __restrict__ Cf, u16* __restrict__ Cb,
                               u64 ldc, int M, int N, int relu)
{
    __shared__ u16 sA[16 * APITCH_];
    const int tid  = threadIdx.x;
    const int lane = tid & 31;
    const int wave = tid >> 5;
    const int nt = (N + 15) >> 4;
    int tn = blockIdx.x * 4 + wave;
    if (tn >= nt) tn = nt - 1;            // tail waves recompute last tile (keeps barriers full)
    const int tm = blockIdx.y;
    const int rl = lane & 15;
    const int kh = lane >> 4;             // which K-half this lane holds
    const int n  = tn * 16 + rl;          // W is padded to the tile grid

    const u16* Wr  = W + (u64)n * (u64)Kp + kh * 8;
    // async-copy mapping: 128 threads x 16B = one 16x64 bf16 A tile per chunk
    const int crow = tid >> 3, cseg = tid & 7;
    const u16* Ac  = A + (u64)(tm * 16 + crow) * strideA + cseg * 8;
    const unsigned ldsdst = (unsigned)(u64)&sA[crow * APITCH_ + cseg * 8];
    const u16* sAr = &sA[rl * APITCH_ + kh * 8];

    v8f acc = {0.f, 0.f, 0.f, 0.f, 0.f, 0.f, 0.f, 0.f};
    for (int k0 = 0; k0 < Kp; k0 += KC_) {
        __syncthreads();                  // previous chunk fully consumed
        asm volatile("global_load_async_to_lds_b128 %0, %1, off"
                     :: "v"(ldsdst), "v"((u64)(Ac + k0)) : "memory");
        if (k0 + KC_ < Kp)                // gfx1250 global_prefetch path for W stream
            __builtin_prefetch(Wr + k0 + KC_, 0, 3);
        asm volatile("s_wait_asynccnt 0x0" ::: "memory");
        __syncthreads();                  // chunk visible to all 4 waves
#pragma unroll
        for (int ks = 0; ks < 2; ++ks) {
            Frag a, b;
            a.q[0] = *(const uint4*)(sAr + ks * 32);        // ds_load_b128
            a.q[1] = *(const uint4*)(sAr + ks * 32 + 16);
            b.q[0] = *(const uint4*)(Wr + k0 + ks * 32);    // global_load_b128
            b.q[1] = *(const uint4*)(Wr + k0 + ks * 32 + 16);
            acc = __builtin_amdgcn_wmma_f32_16x16x32_bf16(false, a.v, false, b.v,
                                                          (short)0, acc, false, false);
        }
    }
    int col = tn * 16 + rl;
    if (col < N) {
        float bv = bias ? bias[col] : 0.f;
#pragma unroll
        for (int g = 0; g < 8; ++g) {
            int row = tm * 16 + g + 8 * kh;   // C/D layout: VGPR g -> M = g + 8*(lane/16)
            if (row < M) {
                float v = acc[g] + bv;
                if (relu) v = fmaxf(v, 0.f);
                u64 o = (u64)row * ldc + col;
                if (Cf) Cf[o] = v;
                if (Cb) Cb[o] = f2bf(v);
            }
        }
    }
}

// ---------- RNN recurrent step: h = tanh(xp[t] + h_prev @ Whh^T + bhh) ----------
// hprev/hcur: bf16 [2][M][Hd]; Whh: bf16 [2][Hd][Hd]; bhh: f32 [2][Hd];
// xp: f32 [2][M][T][Hd]; seqout: bf16 [M][T][2*Hd] (dir d at column offset d*Hd).
// Small K, no barriers -> direct global loads.
__global__ void k_rnn_step(const u16* __restrict__ hprev, const u16* __restrict__ Whh,
                           const float* __restrict__ bhh, const float* __restrict__ xp,
                           u16* __restrict__ hcur, u16* __restrict__ seqout,
                           int M, int Hd, int T, int s)
{
    int lane = threadIdx.x & 31;
    int wave = threadIdx.x >> 5;
    int tn = blockIdx.x * 4 + wave;
    if (tn * 16 >= Hd) return;
    int tm = blockIdx.y;
    int d  = blockIdx.z;
    int tt = d ? (T - 1 - s) : s;
    int rl = lane & 15;
    int kh = lane >> 4;
    int r  = tm * 16 + rl; if (r >= M) r = M - 1;
    const u16* Ar = hprev + ((u64)d * M + r) * Hd + kh * 8;
    const u16* Wr = Whh   + ((u64)d * Hd + (tn * 16 + rl)) * Hd + kh * 8;

    v8f acc = {0.f, 0.f, 0.f, 0.f, 0.f, 0.f, 0.f, 0.f};
    for (int k0 = 0; k0 < Hd; k0 += 32) {
        Frag a, b;
        a.q[0] = *(const uint4*)(Ar + k0);
        a.q[1] = *(const uint4*)(Ar + k0 + 16);
        b.q[0] = *(const uint4*)(Wr + k0);
        b.q[1] = *(const uint4*)(Wr + k0 + 16);
        acc = __builtin_amdgcn_wmma_f32_16x16x32_bf16(false, a.v, false, b.v,
                                                      (short)0, acc, false, false);
    }
    int col = tn * 16 + rl;
    float bv = bhh[d * Hd + col];
#pragma unroll
    for (int g = 0; g < 8; ++g) {
        int row = tm * 16 + g + 8 * kh;
        if (row < M) {
            float v = tanhf(acc[g] + bv + xp[(((u64)d * M + row) * T + tt) * Hd + col]);
            u16 h = f2bf(v);
            hcur[((u64)d * M + row) * Hd + col] = h;
            seqout[((u64)row * T + tt) * (u64)(2 * Hd) + (u64)d * Hd + col] = h;
        }
    }
}

// ---------- elementwise / packing kernels ----------
__global__ void k_f32_to_bf16(const float* __restrict__ s, u16* __restrict__ d, u64 n) {
    for (u64 i = blockIdx.x * 256ull + threadIdx.x; i < n; i += (u64)gridDim.x * 256ull)
        d[i] = f2bf(s[i]);
}
__global__ void k_fillz16(u16* __restrict__ d, u64 n) {
    for (u64 i = blockIdx.x * 256ull + threadIdx.x; i < n; i += (u64)gridDim.x * 256ull)
        d[i] = 0;
}
// W [N][K] f32 -> [Npad][Kpad] bf16, zero padded
__global__ void k_conv_nk(const float* __restrict__ s, u16* __restrict__ d,
                          int N, int K, int Npad, int Kpad) {
    u64 n = (u64)Npad * Kpad;
    for (u64 i = blockIdx.x * 256ull + threadIdx.x; i < n; i += (u64)gridDim.x * 256ull) {
        int k = (int)(i % Kpad); int nn = (int)(i / Kpad);
        d[i] = (nn < N && k < K) ? f2bf(s[(u64)nn * K + k]) : (u16)0;
    }
}
// W [K][N] f32 -> transposed [Npad][Kpad] bf16, zero padded
__global__ void k_conv_kn(const float* __restrict__ s, u16* __restrict__ d,
                          int N, int K, int Npad, int Kpad) {
    u64 n = (u64)Npad * Kpad;
    for (u64 i = blockIdx.x * 256ull + threadIdx.x; i < n; i += (u64)gridDim.x * 256ull) {
        int k = (int)(i % Kpad); int nn = (int)(i / Kpad);
        d[i] = (nn < N && k < K) ? f2bf(s[(u64)k * N + nn]) : (u16)0;
    }
}
// res_features [B,6,T,D] f32 -> rf_h [B*T][D] bf16 (node 0) + rf_o [B*5*T][D] bf16 (nodes 1..5)
__global__ void k_conv_rf(const float* __restrict__ rf, u16* __restrict__ rf_h,
                          u16* __restrict__ rf_o, u64 n) {
    for (u64 i = blockIdx.x * 256ull + threadIdx.x; i < n; i += (u64)gridDim.x * 256ull) {
        int dD = (int)(i % D_);
        u64 r = i / D_;
        int t = (int)(r % T_); r /= T_;
        int v = (int)(r % 6);
        u64 b = r / 6;
        u16 val = f2bf(rf[i]);
        if (v == 0) rf_h[(b * T_ + t) * (u64)D_ + dD] = val;
        else        rf_o[((b * NOBJ_ + (v - 1)) * T_ + t) * (u64)D_ + dD] = val;
    }
}
__device__ __forceinline__ float obj_node(const float* bbox, const float* onehot,
                                          const u16* ofeat, u64 b, int t, int v, int c) {
    if (c >= 1004 && c < 2004) return bf2f(ofeat[((b * NOBJ_ + (v - 1)) * T_ + t) * (u64)PP_ + (c - 1004)]);
    if (c >= 2004 && c < 2008) return bbox[((b * 6 + v) * T_ + t) * 4 + (c - 2004)];
    if (c >= 2008 && c < 2018) return onehot[((b * NOBJ_ + (v - 1)) * T_ + t) * 10 + (c - 2008)];
    return 0.f;
}
// gmix[b,t,u,c] = sum_v A_hat[u][v] * node(b,t,v,c); bf16 out [B*T*6][2048]
__global__ void k_build_gmix(const float* __restrict__ bbox, const float* __restrict__ onehot,
                             const u16* __restrict__ hfeat, const u16* __restrict__ ofeat,
                             u16* __restrict__ gmix, u64 n) {
    const float r6 = 1.f / 6.f, r12 = 0.28867513459481288f; // 1/sqrt(12)
    for (u64 i = blockIdx.x * 256ull + threadIdx.x; i < n; i += (u64)gridDim.x * 256ull) {
        int c = (int)(i % INP_);
        u64 r = i / INP_;
        int u = (int)(r % 6); r /= 6;
        int t = (int)(r % T_);
        u64 b = r / T_;
        float n0 = 0.f;
        if (c < 4)          n0 = bbox[((b * 6 + 0) * T_ + t) * 4 + c];
        else if (c < 1004)  n0 = bf2f(hfeat[(b * T_ + t) * (u64)PP_ + (c - 4)]);
        float out;
        if (u == 0) {
            float s = 0.f;
            for (int v = 1; v < 6; ++v) s += obj_node(bbox, onehot, ofeat, b, t, v, c);
            out = r6 * n0 + r12 * s;
        } else {
            out = r12 * n0 + 0.5f * obj_node(bbox, onehot, ofeat, b, t, u, c);
        }
        gmix[i] = f2bf(out);
    }
}
// second graph mix over 6 nodes: gmix2[bt,u,h] = sum_v A_hat[u][v]*g1[bt,v,h]
__global__ void k_mix2(const u16* __restrict__ g1, u16* __restrict__ gmix2, u64 n) {
    const float r6 = 1.f / 6.f, r12 = 0.28867513459481288f;
    for (u64 i = blockIdx.x * 256ull + threadIdx.x; i < n; i += (u64)gridDim.x * 256ull) {
        int h = (int)(i % H_);
        u64 r = i / H_;
        int u = (int)(r % 6);
        u64 bt = r / 6;
        const u16* g = g1 + bt * 6ull * H_ + h;
        float g0 = bf2f(g[0]);
        float out;
        if (u == 0) {
            float s = 0.f;
            for (int v = 1; v < 6; ++v) s += bf2f(g[(u64)v * H_]);
            out = r6 * g0 + r12 * s;
        } else {
            out = r12 * g0 + 0.5f * bf2f(g[(u64)u * H_]);
        }
        gmix2[i] = f2bf(out);
    }
}
// obj RNN input: [pair=b*5+j][t][0:512]=human node feats, [512:1024]=obj node j+1
__global__ void k_objin(const u16* __restrict__ spatial, u16* __restrict__ objin, u64 n) {
    for (u64 i = blockIdx.x * 256ull + threadIdx.x; i < n; i += (u64)gridDim.x * 256ull) {
        int h = (int)(i % 1024);
        u64 r = i / 1024;
        int t = (int)(r % T_);
        u64 p = r / T_;
        u64 b = p / NOBJ_;
        int j = (int)(p % NOBJ_);
        int node = (h < H_) ? 0 : (j + 1);
        objin[i] = spatial[((b * T_ + t) * 6 + node) * (u64)H_ + (h & (H_ - 1))];
    }
}
__global__ void k_prefix(const int* __restrict__ num_objs, int* __restrict__ pre) {
    if (threadIdx.x == 0 && blockIdx.x == 0) {
        int s = 0;
        for (int b = 0; b < B_; ++b) { pre[b] = s; s += num_objs[b]; }
        pre[B_] = s;
    }
}
__global__ void k_subact(const float* __restrict__ hh2, float* __restrict__ out) {
    int i = blockIdx.x * 256 + threadIdx.x;
    if (i >= B_ * 10) return;
    int b = i / 10, c = i % 10;
    float s = 0.f;
    for (int t = 0; t < T_; ++t) s += hh2[((u64)b * T_ + t) * 16 + c];
    out[b * 10 + c] = s;
}
__global__ void k_afford(const float* __restrict__ oh2, const int* __restrict__ num_objs,
                         const int* __restrict__ pre, float* __restrict__ out) {
    int i = blockIdx.x * 256 + threadIdx.x;
    if (i >= B_ * NOBJ_ * 12) return;
    int c = i % 12;
    int p = i / 12;
    int b = p / NOBJ_, j = p % NOBJ_;
    if (j >= num_objs[b]) return;
    float s = 0.f;
    for (int t = 0; t < T_; ++t) s += oh2[((u64)p * T_ + t) * 16 + c];
    out[(u64)B_ * 10 + ((u64)(pre[b] + j)) * 12 + c] = s;
}

// ---------- host-side launch helpers ----------
static inline void gemm(hipStream_t st, const u16* A, u64 strideA, const u16* W, int Kp,
                        const float* bias, float* Cf, u16* Cb, u64 ldc, int M, int N, int relu) {
    int nt = (N + 15) / 16, mt = (M + 15) / 16;
    dim3 grid((nt + 3) / 4, mt, 1), blk(128, 1, 1);
    wmma_gemm_bf16<<<grid, blk, 0, st>>>(A, strideA, W, Kp, bias, Cf, Cb, ldc, M, N, relu);
}
static inline dim3 ewgrid(u64 n) { return dim3((unsigned)((n + 255) / 256), 1, 1); }

extern "C" void kernel_launch(void* const* d_in, const int* in_sizes, int n_in,
                              void* d_out, int out_size, void* d_ws, size_t ws_size,
                              hipStream_t stream) {
    (void)in_sizes; (void)n_in; (void)out_size; (void)ws_size;
    // inputs (setup_inputs order)
    const float* one_hot  = (const float*)d_in[0];
    const float* bboxes   = (const float*)d_in[1];
    const int*   num_objs = (const int*)d_in[2];
    const float* rf       = (const float*)d_in[3];
    const float* Wph = (const float*)d_in[4];  const float* bph = (const float*)d_in[5];
    const float* Wpo = (const float*)d_in[6];  const float* bpo = (const float*)d_in[7];
    const float* Wg1 = (const float*)d_in[8];  const float* bg1 = (const float*)d_in[9];
    const float* Wg2 = (const float*)d_in[10]; const float* bg2 = (const float*)d_in[11];
    const float* sub_Wih = (const float*)d_in[12]; const float* sub_Whh = (const float*)d_in[13];
    const float* sub_bih = (const float*)d_in[14]; const float* sub_bhh = (const float*)d_in[15];
    const float* aff_Wih = (const float*)d_in[16]; const float* aff_Whh = (const float*)d_in[17];
    const float* aff_bih = (const float*)d_in[18]; const float* aff_bhh = (const float*)d_in[19];
    const float* ch_W1 = (const float*)d_in[20]; const float* ch_b1 = (const float*)d_in[21];
    const float* ch_W2 = (const float*)d_in[22]; const float* ch_b2 = (const float*)d_in[23];
    const float* co_W1 = (const float*)d_in[24]; const float* co_b1 = (const float*)d_in[25];
    const float* co_W2 = (const float*)d_in[26]; const float* co_b2 = (const float*)d_in[27];
    float* out = (float*)d_out;

    // deterministic bump allocator over d_ws
    char* base = (char*)d_ws;
    u64 off = 0;
    auto alloc = [&](u64 bytes) -> void* {
        void* r = base + off;
        off = (off + bytes + 255ull) & ~255ull;
        return r;
    };
    const int NPAIR = B_ * NOBJ_;              // 1280 (all pairs; compacted at the end)
    u16* rf_h   = (u16*)alloc((u64)B_ * T_ * D_ * 2);
    u16* rf_o   = (u16*)alloc((u64)NPAIR * T_ * D_ * 2);
    u16* wph    = (u16*)alloc((u64)PP_ * D_ * 2);
    u16* wpo    = (u16*)alloc((u64)PP_ * D_ * 2);
    u16* wg1    = (u16*)alloc((u64)H_ * INP_ * 2);
    u16* wg2    = (u16*)alloc((u64)H_ * H_ * 2);
    u16* swih   = (u16*)alloc((u64)2 * 2 * HH_ * H_ * 2);
    u16* swhh   = (u16*)alloc((u64)2 * 2 * HH_ * HH_ * 2);
    u16* awih   = (u16*)alloc((u64)2 * 2 * H_ * (2 * H_) * 2);
    u16* awhh   = (u16*)alloc((u64)2 * 2 * H_ * H_ * 2);
    u16* chw1   = (u16*)alloc((u64)H_ * H_ * 2);
    u16* chw2   = (u16*)alloc((u64)16 * H_ * 2);
    u16* cow1   = (u16*)alloc((u64)H_ * (2 * H_) * 2);
    u16* cow2   = (u16*)alloc((u64)16 * H_ * 2);
    u16* hfeat  = (u16*)alloc((u64)B_ * T_ * PP_ * 2);
    u16* ofeat  = (u16*)alloc((u64)NPAIR * T_ * PP_ * 2);
    u16* gmix   = (u16*)alloc((u64)B_ * T_ * 6 * INP_ * 2);
    u16* g1b    = (u16*)alloc((u64)B_ * T_ * 6 * H_ * 2);
    u16* gmix2  = (u16*)alloc((u64)B_ * T_ * 6 * H_ * 2);
    u16* spatial= (u16*)alloc((u64)B_ * T_ * 6 * H_ * 2);
    u16* objin  = (u16*)alloc((u64)NPAIR * T_ * 2 * H_ * 2);
    float* xph  = (float*)alloc((u64)2 * B_ * T_ * HH_ * 4);
    float* xpo  = (float*)alloc((u64)2 * NPAIR * T_ * H_ * 4);
    u16* hA     = (u16*)alloc((u64)2 * NPAIR * H_ * 2);
    u16* hB     = (u16*)alloc((u64)2 * NPAIR * H_ * 2);
    u16* hout1  = (u16*)alloc((u64)B_ * T_ * H_ * 2);
    u16* hout2  = (u16*)alloc((u64)B_ * T_ * H_ * 2);
    u16* oout1  = (u16*)alloc((u64)NPAIR * T_ * 2 * H_ * 2);
    u16* oout2  = (u16*)alloc((u64)NPAIR * T_ * 2 * H_ * 2);
    u16* hh1    = (u16*)alloc((u64)B_ * T_ * H_ * 2);
    float* hh2  = (float*)alloc((u64)B_ * T_ * 16 * 4);
    u16* oh1    = (u16*)alloc((u64)NPAIR * T_ * H_ * 2);
    float* oh2  = (float*)alloc((u64)NPAIR * T_ * 16 * 4);
    int* pre    = (int*)alloc((u64)(B_ + 1) * 4);

    // ---- pack inputs/weights to bf16 ----
    u64 nrf = (u64)B_ * 6 * T_ * D_;
    k_conv_rf<<<ewgrid(nrf), 256, 0, stream>>>(rf, rf_h, rf_o, nrf);
    k_conv_nk<<<ewgrid((u64)PP_ * D_), 256, 0, stream>>>(Wph, wph, P_, D_, PP_, D_);
    k_conv_nk<<<ewgrid((u64)PP_ * D_), 256, 0, stream>>>(Wpo, wpo, P_, D_, PP_, D_);
    k_conv_kn<<<ewgrid((u64)H_ * INP_), 256, 0, stream>>>(Wg1, wg1, H_, 2018, H_, INP_);
    k_conv_kn<<<ewgrid((u64)H_ * H_), 256, 0, stream>>>(Wg2, wg2, H_, H_, H_, H_);
    k_f32_to_bf16<<<ewgrid((u64)4 * HH_ * H_), 256, 0, stream>>>(sub_Wih, swih, (u64)4 * HH_ * H_);
    k_f32_to_bf16<<<ewgrid((u64)4 * HH_ * HH_), 256, 0, stream>>>(sub_Whh, swhh, (u64)4 * HH_ * HH_);
    k_f32_to_bf16<<<ewgrid((u64)4 * H_ * 2 * H_), 256, 0, stream>>>(aff_Wih, awih, (u64)4 * H_ * 2 * H_);
    k_f32_to_bf16<<<ewgrid((u64)4 * H_ * H_), 256, 0, stream>>>(aff_Whh, awhh, (u64)4 * H_ * H_);
    k_f32_to_bf16<<<ewgrid((u64)H_ * H_), 256, 0, stream>>>(ch_W1, chw1, (u64)H_ * H_);
    k_conv_nk<<<ewgrid((u64)16 * H_), 256, 0, stream>>>(ch_W2, chw2, 10, H_, 16, H_);
    k_f32_to_bf16<<<ewgrid((u64)H_ * 2 * H_), 256, 0, stream>>>(co_W1, cow1, (u64)H_ * 2 * H_);
    k_conv_nk<<<ewgrid((u64)16 * H_), 256, 0, stream>>>(co_W2, cow2, 12, H_, 16, H_);

    // ---- feature projections (the two biggest GEMMs) ----
    gemm(stream, rf_h, D_, wph, D_, bph, nullptr, hfeat, PP_, B_ * T_, P_, 0);
    gemm(stream, rf_o, D_, wpo, D_, bpo, nullptr, ofeat, PP_, NPAIR * T_, P_, 0);

    // ---- graph conv ----
    u64 ngm = (u64)B_ * T_ * 6 * INP_;
    k_build_gmix<<<ewgrid(ngm), 256, 0, stream>>>(bboxes, one_hot, hfeat, ofeat, gmix, ngm);
    gemm(stream, gmix, INP_, wg1, INP_, bg1, nullptr, g1b, H_, B_ * T_ * 6, H_, 1);
    u64 nm2 = (u64)B_ * T_ * 6 * H_;
    k_mix2<<<ewgrid(nm2), 256, 0, stream>>>(g1b, gmix2, nm2);
    gemm(stream, gmix2, H_, wg2, H_, bg2, nullptr, spatial, H_, B_ * T_ * 6, H_, 1);

    // ---- obj RNN input (all B*NOBJ pairs; masked at the end) ----
    u64 noi = (u64)NPAIR * T_ * 2 * H_;
    k_objin<<<ewgrid(noi), 256, 0, stream>>>(spatial, objin, noi);

    // ---- human RNN: 2 layers, bidirectional, Hd=256 ----
    {
        const int M = B_, Hd = HH_;
        for (int l = 0; l < 2; ++l) {
            const u16* lin = (l == 0) ? spatial : hout1;
            u64 strideA = (l == 0) ? (u64)6 * H_ : (u64)2 * Hd;  // node-0 view / concat out
            int Kin = (l == 0) ? H_ : 2 * Hd;
            for (int d = 0; d < 2; ++d)
                gemm(stream, lin, strideA, swih + (u64)(l * 2 + d) * Hd * Kin, Kin,
                     sub_bih + (l * 2 + d) * Hd, xph + (u64)d * M * T_ * Hd, nullptr,
                     Hd, M * T_, Hd, 0);
            k_fillz16<<<ewgrid((u64)2 * M * Hd), 256, 0, stream>>>(hA, (u64)2 * M * Hd);
            u16* seq = (l == 0) ? hout1 : hout2;
            u16* cur = hA; u16* nxt = hB;
            dim3 grid((Hd / 16 + 3) / 4, M / 16, 2), blk(128, 1, 1);
            for (int s = 0; s < T_; ++s) {
                k_rnn_step<<<grid, blk, 0, stream>>>(cur, swhh + (u64)l * 2 * Hd * Hd,
                                                     sub_bhh + l * 2 * Hd, xph, nxt, seq,
                                                     M, Hd, T_, s);
                u16* t = cur; cur = nxt; nxt = t;
            }
        }
    }

    // ---- obj RNN: 2 layers, bidirectional, Hd=512, M=1280 pairs ----
    {
        const int M = NPAIR, Hd = H_;
        for (int l = 0; l < 2; ++l) {
            const u16* lin = (l == 0) ? objin : oout1;
            int Kin = 2 * Hd;                               // both layers take 1024-wide input
            for (int d = 0; d < 2; ++d)
                gemm(stream, lin, (u64)Kin, awih + (u64)(l * 2 + d) * Hd * Kin, Kin,
                     aff_bih + (l * 2 + d) * Hd, xpo + (u64)d * M * T_ * Hd, nullptr,
                     Hd, M * T_, Hd, 0);
            k_fillz16<<<ewgrid((u64)2 * M * Hd), 256, 0, stream>>>(hA, (u64)2 * M * Hd);
            u16* seq = (l == 0) ? oout1 : oout2;
            u16* cur = hA; u16* nxt = hB;
            dim3 grid((Hd / 16 + 3) / 4, M / 16, 2), blk(128, 1, 1);
            for (int s = 0; s < T_; ++s) {
                k_rnn_step<<<grid, blk, 0, stream>>>(cur, awhh + (u64)l * 2 * Hd * Hd,
                                                     aff_bhh + l * 2 * Hd, xpo, nxt, seq,
                                                     M, Hd, T_, s);
                u16* t = cur; cur = nxt; nxt = t;
            }
        }
    }

    // ---- heads + time reduction + compaction ----
    gemm(stream, hout2, (u64)H_, chw1, H_, ch_b1, nullptr, hh1, H_, B_ * T_, H_, 1);
    gemm(stream, hh1, (u64)H_, chw2, H_, ch_b2, hh2, nullptr, 16, B_ * T_, 10, 0);
    gemm(stream, oout2, (u64)2 * H_, cow1, 2 * H_, co_b1, nullptr, oh1, H_, NPAIR * T_, H_, 1);
    gemm(stream, oh1, (u64)H_, cow2, H_, co_b2, oh2, nullptr, 16, NPAIR * T_, 12, 0);
    k_prefix<<<1, 32, 0, stream>>>(num_objs, pre);
    k_subact<<<(B_ * 10 + 255) / 256, 256, 0, stream>>>(hh2, out);
    k_afford<<<(B_ * NOBJ_ * 12 + 255) / 256, 256, 0, stream>>>(oh2, num_objs, pre, out);
}
